// Qwen3_5MoeGatedDeltaNetTRTNative_30855045055137
// MI455X (gfx1250) — matrix operand order repaired and checked by reading, not compiled, not run on hardware
//
#include <hip/hip_runtime.h>
#include <hip/hip_bf16.h>

#define H_      16
#define DK_     128
#define DV_     128
#define HID_    3072
#define KD_     2048
#define VD_     2048
#define QKVZ_N  8192      // 2*KD + 2*VD
#define MIX_N   6144      // 2*KD + VD
#define T_      4096
#define B_      2
#define ROWS    8192      // B*T
#define EPS_    1e-6f

typedef __bf16 bf16_t;
typedef __attribute__((ext_vector_type(16))) __bf16 bf16x16;
typedef __attribute__((ext_vector_type(8)))  float  f32x8;

__device__ __forceinline__ bf16_t f2bf(float x) {
  unsigned u = __float_as_uint(x);
  u += 0x7FFFu + ((u >> 16) & 1u);           // round-to-nearest-even
  unsigned short h = (unsigned short)(u >> 16);
  bf16_t r;
  __builtin_memcpy(&r, &h, 2);
  return r;
}

union FragCast {
  struct { uint4 lo, hi; } u;
  bf16x16 v;
};

__device__ __forceinline__ bf16x16 ldFrag(const bf16_t* p) {
  FragCast f;
  f.u.lo = *(const uint4*)p;
  f.u.hi = *(const uint4*)(p + 8);
  return f.v;
}

// ---------------------------------------------------------------- f32 -> bf16
__global__ __launch_bounds__(256) void cvt_f32_bf16(const float* __restrict__ src,
                                                    bf16_t* __restrict__ dst,
                                                    long long n) {
  long long i = (long long)blockIdx.x * blockDim.x + threadIdx.x;
  long long stride = (long long)gridDim.x * blockDim.x;
  for (; i < n; i += stride) dst[i] = f2bf(src[i]);
}

// ---------------------------------------------------------------- bf16 WMMA GEMM
// C[M,N] = A[M,K] * B[K,N], row-major. Block tile 128x128, K-tile 64.
// 8 waves as 4(M) x 2(N); wave tile 32x64 = 2x4 WMMA fragments.
// A tile: global_load_async_to_lds_b128 (ASYNCcnt) straight into a padded
// LDS layout (no VGPR staging). B tile: register-staged transposed store.
// Double-buffered LDS, one barrier per K-tile.
#define BMT 128
#define BNT 128
#define BKT 64
#define LDT 72          // padded LDS row stride (bf16 elems)

template<int N, int K>
__global__ __launch_bounds__(256, 1) void gemm_bf16_wmma(
    const bf16_t* __restrict__ A, const bf16_t* __restrict__ Bm,
    float* __restrict__ C)
{
  __shared__ __align__(16) bf16_t As[2][BMT][LDT];
  __shared__ __align__(16) bf16_t Bs[2][BNT][LDT];

  const int tid  = threadIdx.x;
  const int lane = tid & 31;
  const int wave = tid >> 5;     // 0..7
  const int wm   = wave >> 1;    // 0..3 (M direction, 32 rows each)
  const int wn   = wave & 1;     // 0..1 (N direction, 64 cols each)
  const int m0   = blockIdx.y * BMT;
  const int n0   = blockIdx.x * BNT;
  const int lh   = lane >> 4;    // lane half
  const int ll   = lane & 15;

  f32x8 acc[2][4] = {};

  // A: thread covers half a row (32 K) => 4 async 16B chunks
  const int arow  = tid >> 1;          // 0..127
  const int ahalf = tid & 1;           // which 32-K half

  uint4 bReg[4];

  // Async DMA of one A tile into LDS in contiguous K order
  // (no group swizzle) at the per-lane LDS destination.
  auto asyncA = [&](int k0, int buf) {
    #pragma unroll
    for (int q = 0; q < 4; ++q) {
      int qs = q;
      const bf16_t* g = A + (size_t)(m0 + arow) * K + k0 + ahalf * 32 + q * 8;
      unsigned l = (unsigned)(size_t)&As[buf][arow][ahalf * 32 + qs * 8];
      asm volatile("global_load_async_to_lds_b128 %0, %1, off"
                   :: "v"(l), "v"(g) : "memory");
    }
  };
  auto loadB = [&](int k0) {
    #pragma unroll
    for (int i = 0; i < 4; ++i) {
      int c  = tid + 256 * i;          // 0..1023
      int kk = c >> 4;                 // 0..63
      int nb = (c & 15) * 8;           // 0..120
      bReg[i] = *(const uint4*)(Bm + (size_t)(k0 + kk) * N + n0 + nb);
    }
  };
  auto storeB = [&](int buf) {
    #pragma unroll
    for (int i = 0; i < 4; ++i) {
      int c  = tid + 256 * i;
      int kk = c >> 4;
      int nb = (c & 15) * 8;
      bf16_t tmp[8];
      *(uint4*)tmp = bReg[i];
      #pragma unroll
      for (int e = 0; e < 8; ++e) Bs[buf][nb + e][kk] = tmp[e];
    }
  };
  auto compute = [&](int buf) {
    #pragma unroll
    for (int ks = 0; ks < 2; ++ks) {
      bf16x16 af[2], bfr[4];
      #pragma unroll
      for (int mi = 0; mi < 2; ++mi)
        af[mi] = ldFrag(&As[buf][wm * 32 + mi * 16 + ll][ks * 32 + lh * 16]);
      #pragma unroll
      for (int ni = 0; ni < 4; ++ni)
        bfr[ni] = ldFrag(&Bs[buf][wn * 64 + ni * 16 + ll][ks * 32 + lh * 16]);
      #pragma unroll
      for (int mi = 0; mi < 2; ++mi)
        #pragma unroll
        for (int ni = 0; ni < 4; ++ni)
          acc[mi][ni] = __builtin_amdgcn_wmma_f32_16x16x32_bf16(
              false, af[mi], false, bfr[ni],
              (short)0, acc[mi][ni], false, false);
    }
  };

  // ---- prologue: tile 0 into buffer 0
  asyncA(0, 0);
  loadB(0);
  storeB(0);
  asm volatile("s_wait_asynccnt 0" ::: "memory");
  __syncthreads();

  int buf = 0;
  for (int k0 = 0; k0 < K; k0 += BKT) {
    const int nbuf = buf ^ 1;
    const bool hasNext = (k0 + BKT) < K;
    if (hasNext) {
      asyncA(k0 + BKT, nbuf);      // DMA next A tile while we compute
      loadB(k0 + BKT);             // issue next B loads before compute
      if (k0 + 2 * BKT < K)        // keep L2 one more tile ahead
        __builtin_prefetch(Bm + (size_t)(k0 + 2 * BKT + (tid >> 4)) * N + n0 + (tid & 15) * 8, 0, 2);
    }
    compute(buf);                  // 16 WMMAs overlap async DMA + B loads
    if (hasNext) storeB(nbuf);     // loadcnt wait lands here, post-compute
    asm volatile("s_wait_asynccnt 0" ::: "memory");
    __syncthreads();
    buf = nbuf;
  }

  // ---- epilogue: N is constexpr so row offsets fold into immediates
  #pragma unroll
  for (int mi = 0; mi < 2; ++mi)
    #pragma unroll
    for (int ni = 0; ni < 4; ++ni) {
      int nn = n0 + wn * 64 + ni * 16 + ll;
      int mb = m0 + wm * 32 + mi * 16 + lh * 8;
      float* cp = C + (size_t)mb * N + nn;
      #pragma unroll
      for (int r = 0; r < 8; ++r) cp[(size_t)r * N] = acc[mi][ni][r];
    }
}

// ---------------------------------------------------------------- ba -> g, beta
__global__ __launch_bounds__(32) void ba_kernel(
    const float* __restrict__ hs, const float* __restrict__ Wba,
    const float* __restrict__ dt_bias, const float* __restrict__ A_log,
    float* __restrict__ gbuf, float* __restrict__ bbuf)
{
  int row = blockIdx.x;
  int j   = threadIdx.x;       // 0..31
  const float* hr = hs + (size_t)row * HID_;
  float acc = 0.f;
  for (int k = 0; k < HID_; ++k) acc = fmaf(hr[k], Wba[k * 32 + j], acc);
  if (j < H_) {
    bbuf[row * H_ + j] = 1.f / (1.f + __expf(-acc));
  } else {
    int h = j - H_;
    float x  = acc + dt_bias[h];
    float sp = (x > 0.f) ? (x + log1pf(__expf(-x))) : log1pf(__expf(x));
    gbuf[row * H_ + h] = -__expf(A_log[h]) * sp;
  }
}

// ---------------------------------------------------------------- causal conv + silu
#define CT 32
__global__ __launch_bounds__(256) void conv_silu_kernel(
    const float* __restrict__ qkvz, const float* __restrict__ conv_w,
    const float* __restrict__ conv_b, float* __restrict__ qkv)
{
  __shared__ float sm[CT + 3][128];
  int c0 = blockIdx.x * 128;
  int t0 = blockIdx.y * CT;
  int b  = blockIdx.z;
  for (int idx = threadIdx.x; idx < (CT + 3) * 128; idx += 256) {
    int tt = idx >> 7;
    int c  = idx & 127;
    int t  = t0 + tt - 3;
    sm[tt][c] = (t >= 0) ? qkvz[(size_t)(b * T_ + t) * QKVZ_N + c0 + c] : 0.f;
  }
  __syncthreads();
  int c  = threadIdx.x & 127;
  int ts = threadIdx.x >> 7;   // 0..1
  float w0 = conv_w[(c0 + c) * 4 + 0];
  float w1 = conv_w[(c0 + c) * 4 + 1];
  float w2 = conv_w[(c0 + c) * 4 + 2];
  float w3 = conv_w[(c0 + c) * 4 + 3];
  float bias = conv_b[c0 + c];
  for (int tt = ts; tt < CT; tt += 2) {
    float acc = bias + w0 * sm[tt][c] + w1 * sm[tt + 1][c]
                     + w2 * sm[tt + 2][c] + w3 * sm[tt + 3][c];
    float s = acc / (1.f + __expf(-acc));   // silu
    qkv[(size_t)(b * T_ + t0 + tt) * MIX_N + c0 + c] = s;
  }
}

// ---------------------------------------------------------------- l2norm(q,k) (+ q scale)
__global__ __launch_bounds__(128) void l2norm_kernel(float* __restrict__ qkv)
{
  __shared__ float red[4];
  int row = blockIdx.x;
  int hh  = blockIdx.y;        // 0..15 q heads, 16..31 k heads
  size_t base = (size_t)row * MIX_N + ((hh < H_) ? hh * DK_ : KD_ + (hh - H_) * DK_);
  float v  = qkv[base + threadIdx.x];
  float ss = v * v;
  for (int off = 16; off > 0; off >>= 1) ss += __shfl_down(ss, off, 32);
  if ((threadIdx.x & 31) == 0) red[threadIdx.x >> 5] = ss;
  __syncthreads();
  float tot   = red[0] + red[1] + red[2] + red[3];
  float scale = rsqrtf(tot + EPS_);
  if (hh < H_) scale *= 0.08838834764831845f;  // DK^-0.5
  qkv[base + threadIdx.x] = v * scale;
}

// ---------------------------------------------------------------- gated delta scan
// One block per (b,h); state column S[:,v] lives in thread v's registers.
__global__ __launch_bounds__(128) void scan_kernel(
    const float* __restrict__ qkv, const float* __restrict__ gbuf,
    const float* __restrict__ bbuf, float* __restrict__ obuf)
{
  int b = blockIdx.x >> 4;
  int h = blockIdx.x & 15;
  int tid = threadIdx.x;       // v index
  float S[DK_];
  #pragma unroll
  for (int i = 0; i < DK_; ++i) S[i] = 0.f;

  __shared__ float qs[128], ks[128], vs[128];
  __shared__ float gb[2];

  for (int t = 0; t < T_; ++t) {
    int row = b * T_ + t;
    size_t base = (size_t)row * MIX_N + h * DK_;
    qs[tid] = qkv[base + tid];
    ks[tid] = qkv[base + KD_ + tid];
    vs[tid] = qkv[base + 2 * KD_ + tid];
    if (tid == 0) {
      gb[0] = __expf(gbuf[row * H_ + h]);
      gb[1] = bbuf[row * H_ + h];
    }
    __syncthreads();
    float eg = gb[0], bt = gb[1];
    float kv = 0.f;
    #pragma unroll
    for (int i = 0; i < DK_; ++i) { S[i] *= eg; kv = fmaf(ks[i], S[i], kv); }
    float delta = (vs[tid] - kv) * bt;
    float o = 0.f;
    #pragma unroll
    for (int i = 0; i < DK_; ++i) { S[i] = fmaf(ks[i], delta, S[i]); o = fmaf(qs[i], S[i], o); }
    obuf[(size_t)row * VD_ + h * DV_ + tid] = o;
    __syncthreads();
  }
}

// ---------------------------------------------------------------- gate * silu(z), RMS norm -> bf16
__global__ __launch_bounds__(128) void gate_norm_kernel(
    const float* __restrict__ obuf, const float* __restrict__ qkvz,
    const float* __restrict__ norm_w, bf16_t* __restrict__ xn)
{
  __shared__ float red[4];
  int row = blockIdx.x;
  int h   = blockIdx.y;
  int j   = threadIdx.x;
  float o = obuf[(size_t)row * VD_ + h * DV_ + j];
  float z = qkvz[(size_t)row * QKVZ_N + MIX_N + h * DV_ + j];
  float gate = z / (1.f + __expf(-z));
  float x = o * gate;
  float ss = x * x;
  for (int off = 16; off > 0; off >>= 1) ss += __shfl_down(ss, off, 32);
  if ((j & 31) == 0) red[j >> 5] = ss;
  __syncthreads();
  float ms = (red[0] + red[1] + red[2] + red[3]) * (1.f / DV_);
  float y  = x * rsqrtf(ms + EPS_) * norm_w[j];
  xn[(size_t)row * VD_ + h * DV_ + j] = f2bf(y);
}

// ---------------------------------------------------------------- launch
extern "C" void kernel_launch(void* const* d_in, const int* in_sizes, int n_in,
                              void* d_out, int out_size, void* d_ws, size_t ws_size,
                              hipStream_t stream) {
  const float* hs      = (const float*)d_in[0];
  const float* W_qkvz  = (const float*)d_in[1];
  const float* W_ba    = (const float*)d_in[2];
  const float* conv_w  = (const float*)d_in[3];
  const float* conv_b  = (const float*)d_in[4];
  const float* dt_bias = (const float*)d_in[5];
  const float* A_log   = (const float*)d_in[6];
  const float* norm_w  = (const float*)d_in[7];
  const float* W_out   = (const float*)d_in[8];
  (void)in_sizes; (void)n_in; (void)out_size; (void)ws_size;

  char* wsb = (char*)d_ws;
  size_t off = 0;
  auto alloc = [&](size_t bytes) {
    void* p = wsb + off;
    off += (bytes + 255) & ~(size_t)255;
    return p;
  };
  bf16_t* hsb  = (bf16_t*)alloc((size_t)ROWS * HID_ * 2);
  bf16_t* wqb  = (bf16_t*)alloc((size_t)HID_ * QKVZ_N * 2);
  bf16_t* wob  = (bf16_t*)alloc((size_t)VD_ * HID_ * 2);
  float*  qkvz = (float*) alloc((size_t)ROWS * QKVZ_N * 4);
  float*  qkv  = (float*) alloc((size_t)ROWS * MIX_N * 4);
  float*  gbuf = (float*) alloc((size_t)ROWS * H_ * 4);
  float*  bbuf = (float*) alloc((size_t)ROWS * H_ * 4);
  float*  obuf = (float*) alloc((size_t)ROWS * VD_ * 4);
  bf16_t* xn   = (bf16_t*)alloc((size_t)ROWS * VD_ * 2);

  cvt_f32_bf16<<<2048, 256, 0, stream>>>(hs, hsb, (long long)ROWS * HID_);
  cvt_f32_bf16<<<2048, 256, 0, stream>>>(W_qkvz, wqb, (long long)HID_ * QKVZ_N);
  cvt_f32_bf16<<<1024, 256, 0, stream>>>(W_out, wob, (long long)VD_ * HID_);

  // qkvz = hs @ W_qkvz   (8192 x 8192 x 3072)
  gemm_bf16_wmma<QKVZ_N, HID_><<<dim3(QKVZ_N / BNT, ROWS / BMT), 256, 0, stream>>>(
      hsb, wqb, qkvz);

  ba_kernel<<<ROWS, 32, 0, stream>>>(hs, W_ba, dt_bias, A_log, gbuf, bbuf);

  conv_silu_kernel<<<dim3(MIX_N / 128, T_ / CT, B_), 256, 0, stream>>>(
      qkvz, conv_w, conv_b, qkv);

  l2norm_kernel<<<dim3(ROWS, 2 * H_), 128, 0, stream>>>(qkv);

  scan_kernel<<<B_ * H_, 128, 0, stream>>>(qkv, gbuf, bbuf, obuf);

  gate_norm_kernel<<<dim3(ROWS, H_), 128, 0, stream>>>(obuf, qkvz, norm_w, xn);

  // out = xn @ W_out   (8192 x 3072 x 2048)
  gemm_bf16_wmma<HID_, VD_><<<dim3(HID_ / BNT, ROWS / BMT), 256, 0, stream>>>(
      xn, wob, (float*)d_out);
}
